// FourierKARTLayer_47854525612187
// MI455X (gfx1250) — compile-verified, matrix operand order checked by reading, not run on hardware
//
#include <hip/hip_runtime.h>

// ---------------------------------------------------------------------------
// FourierKART layer for MI455X (gfx1250, wave32, WMMA).
//   Stage 1: angle[bs,q] = X0 @ Wc_w^T + Wc_b + w*t        (f32 WMMA 16x16x4)
//   Stage 2: pack W2[f,d] = {A*cos(Bp); A*sin(Bp)} in f16, pre-permuted into
//            the per-lane B-fragment layout of v_wmma_f32_16x16x32_f16
//   Stage 3: out[bs,d] = F(angle) @ W2, F built in-register via v_sin/v_cos,
//            angles staged in LDS, 4 accumulators per wave (A-frag reuse)
// ---------------------------------------------------------------------------

typedef __attribute__((ext_vector_type(2)))  float    v2f;
typedef __attribute__((ext_vector_type(8)))  float    v8f;
typedef __attribute__((ext_vector_type(16))) _Float16 v16h;

static constexpr int BATCH = 2;
static constexpr int SLEN  = 1024;
static constexpr int DIN   = 256;
static constexpr int DOUT  = 64;
static constexpr int Q     = 128;
static constexpr int KH    = 6;           // harmonics
static constexpr int MROWS = BATCH * SLEN;   // 2048
static constexpr int FPQ   = 2 * KH;         // 12 features per q (6 sin + 6 cos)
static constexpr int KFEAT = Q * FPQ;        // 1536 contraction length
static constexpr int KB    = KFEAT / 32;     // 48 wmma k-steps

// K-index pattern of a 16-bit A/B fragment half `i` for the wave half `hi`
// (ISA 7.12.2, 16-bit A-matrix 16x32): lanes 0-15 hold K {0..7,16..23},
// lanes 16-31 hold K {8..15,24..31}.
__device__ __forceinline__ int frag_k16(int i, int hi) {
    return (i & 7) + ((i >> 3) << 4) + hi * 8;
}

// ---------------------------------------------------------------------------
// Stage 1: angle GEMM.  grid = MROWS/16 = 128 blocks, 256 thr (8 waves).
// Wave w computes the 16x16 tile C[m0:m0+16, 16w:16w+16] with f32 WMMA.
// ---------------------------------------------------------------------------
__global__ __launch_bounds__(256) void k_angle_wmma(
    const float* __restrict__ X0,   // [MROWS, DIN]
    const float* __restrict__ t,    // [BATCH]
    const float* __restrict__ Wcw,  // [Q, DIN]
    const float* __restrict__ Wcb,  // [Q]
    const float* __restrict__ w,    // [Q]
    float* __restrict__ ANG)        // [MROWS, Q]
{
    const int tid  = threadIdx.x;
    const int wave = tid >> 5;
    const int lane = tid & 31;
    const int lm   = lane & 15;
    const int hi   = lane >> 4;          // 0: lanes 0-15, 1: lanes 16-31
    const int m0   = blockIdx.x * 16;
    const int n0   = wave * 16;

    // f32 16x4 A layout: lanes 0-15 -> K0,K1 ; lanes 16-31 -> K2,K3 (M = lane%16)
    const int ksel = hi * 2;
    const float* xrow = X0  + (size_t)(m0 + lm) * DIN + ksel;
    const float* wrow = Wcw + (size_t)(n0 + lm) * DIN + ksel;

    v8f acc = {0.f,0.f,0.f,0.f,0.f,0.f,0.f,0.f};
    #pragma unroll 4
    for (int k = 0; k < DIN; k += 4) {
        v2f a = *(const v2f*)(xrow + k);
        v2f b = *(const v2f*)(wrow + k);
        acc = __builtin_amdgcn_wmma_f32_16x16x4_f32(
            /*neg_a=*/false, a, /*neg_b=*/false, b,
            /*c_mod=*/(short)0, acc, /*reuse_a=*/false, /*reuse_b=*/false);
    }

    const int   q     = n0 + lm;
    const int   batch = m0 >> 10;                    // 1024 rows per batch
    const float shift = Wcb[q] + w[q] * t[batch];

    // C/D layout: VGPR j -> M = j + 8*hi, N = lane%16
    #pragma unroll
    for (int j = 0; j < 8; ++j) {
        const int row = m0 + j + hi * 8;
        ANG[(size_t)row * Q + q] = acc[j] + shift;
    }
}

// ---------------------------------------------------------------------------
// Stage 2: fold A,Bp into f16 weights, permuted into B-fragment order.
// Fragment element (kb, nt, lane, i) holds B[32*kb + frag_k16(i,hi)][16*nt + lane%16]
// where B[f][d]: f = q*12+kk,  kk<6 -> A[d,q,kk]*cos(Bp), kk>=6 -> A[d,q,kk-6]*sin(Bp)
// Stored flat so a lane's fragment is 16 contiguous halves (32B -> b128 loads).
// grid = KFEAT*DOUT/256 = 384 blocks, 256 thr.
// ---------------------------------------------------------------------------
__global__ __launch_bounds__(256) void k_w2_pack(
    const float* __restrict__ A,    // [DOUT, Q, KH]
    const float* __restrict__ Bp,   // [DOUT, Q, KH]
    _Float16* __restrict__ W2P)     // [KB*4*32*16] halves
{
    const int idx  = blockIdx.x * 256 + threadIdx.x;
    const int i    = idx & 15;
    const int lane = (idx >> 4) & 31;
    const int nt   = (idx >> 9) & 3;
    const int kb   = idx >> 11;
    const int hi   = lane >> 4;

    const int kf = kb * 32 + frag_k16(i, hi);
    const int d  = nt * 16 + (lane & 15);
    const int q  = kf / FPQ;
    const int kk = kf - q * FPQ;
    const int ks = (kk < KH) ? kk : kk - KH;

    const int   ai  = (d * Q + q) * KH + ks;
    const float av  = A[ai];
    const float bp  = Bp[ai];
    const float val = (kk < KH) ? av * cosf(bp) : av * sinf(bp);
    W2P[idx] = (_Float16)val;
}

// ---------------------------------------------------------------------------
// Stage 3: Fourier-feature GEMM.  grid = MROWS/64 = 32 blocks, 128 thr (4 waves).
// Block stages 64 angle rows in LDS; wave owns 16 rows, all 64 output cols
// (4 accumulators), builds the A fragment once per k-step via v_sin/v_cos.
// ---------------------------------------------------------------------------
__global__ __launch_bounds__(128) void k_fourier_wmma(
    const float*    __restrict__ ANG,  // [MROWS, Q]
    const _Float16* __restrict__ W2P,  // packed B fragments
    float*          __restrict__ out)  // [MROWS, DOUT]
{
    __shared__ float sAng[64 * Q];     // 32 KB
    const int tid     = threadIdx.x;
    const int rowBase = blockIdx.x * 64;

    for (int r = tid; r < 64 * Q; r += 128)
        sAng[r] = ANG[(size_t)rowBase * Q + r];
    __syncthreads();

    const int wave = tid >> 5;
    const int lane = tid & 31;
    const int lm   = lane & 15;
    const int hi   = lane >> 4;
    const int m0   = rowBase + wave * 16;

    const float* ar  = sAng + (wave * 16 + lm) * Q;   // A-frag row (M = lane%16)
    const v16h*  W2v = (const v16h*)W2P;

    v8f acc0 = {0.f,0.f,0.f,0.f,0.f,0.f,0.f,0.f};
    v8f acc1 = {0.f,0.f,0.f,0.f,0.f,0.f,0.f,0.f};
    v8f acc2 = {0.f,0.f,0.f,0.f,0.f,0.f,0.f,0.f};
    v8f acc3 = {0.f,0.f,0.f,0.f,0.f,0.f,0.f,0.f};

    for (int kb = 0; kb < KB; ++kb) {
        // Build A fragment: F[row, 32*kb + pat] = sin/cos((k+1)*angle)
        v16h a;
        #pragma unroll
        for (int i = 0; i < 16; ++i) {
            const int   kf  = kb * 32 + frag_k16(i, hi);
            const int   q   = kf / FPQ;
            const int   kk  = kf - q * FPQ;
            const float ang = ar[q];
            const float kc  = (float)(((kk < KH) ? kk : kk - KH) + 1);
            const float v   = (kk < KH) ? __sinf(kc * ang) : __cosf(kc * ang);
            a[i] = (_Float16)v;
        }
        const v16h* bbase = W2v + (size_t)kb * 128 + lane;
        const v16h b0 = bbase[0];
        const v16h b1 = bbase[32];
        const v16h b2 = bbase[64];
        const v16h b3 = bbase[96];
        acc0 = __builtin_amdgcn_wmma_f32_16x16x32_f16(false, a, false, b0, (short)0, acc0, false, false);
        acc1 = __builtin_amdgcn_wmma_f32_16x16x32_f16(false, a, false, b1, (short)0, acc1, false, false);
        acc2 = __builtin_amdgcn_wmma_f32_16x16x32_f16(false, a, false, b2, (short)0, acc2, false, false);
        acc3 = __builtin_amdgcn_wmma_f32_16x16x32_f16(false, a, false, b3, (short)0, acc3, false, false);
    }

    #pragma unroll
    for (int j = 0; j < 8; ++j) {
        const int row = m0 + j + hi * 8;
        float* o = out + (size_t)row * DOUT + lm;
        o[0]  = acc0[j];
        o[16] = acc1[j];
        o[32] = acc2[j];
        o[48] = acc3[j];
    }
}

// ---------------------------------------------------------------------------
extern "C" void kernel_launch(void* const* d_in, const int* in_sizes, int n_in,
                              void* d_out, int out_size, void* d_ws, size_t ws_size,
                              hipStream_t stream) {
    const float* X0  = (const float*)d_in[0];
    const float* t   = (const float*)d_in[1];
    const float* Wcw = (const float*)d_in[2];
    const float* Wcb = (const float*)d_in[3];
    const float* w   = (const float*)d_in[4];
    const float* A   = (const float*)d_in[5];
    const float* Bp  = (const float*)d_in[6];
    float* out = (float*)d_out;

    char* ws = (char*)d_ws;
    float*    ANG = (float*)ws;                                   // 2048*128*4 = 1 MB
    _Float16* W2P = (_Float16*)(ws + (size_t)MROWS * Q * 4);      // 1536*64*2 = 192 KB

    k_w2_pack   <<<(KFEAT * DOUT) / 256, 256, 0, stream>>>(A, Bp, W2P);
    k_angle_wmma<<<MROWS / 16,           256, 0, stream>>>(X0, t, Wcw, Wcb, w, ANG);
    k_fourier_wmma<<<MROWS / 64,         128, 0, stream>>>(ANG, W2P, out);
}
